// MASGNN_27754078667623
// MI455X (gfx1250) — matrix operand order repaired and checked by reading, not compiled, not run on hardware
//
#include <hip/hip_runtime.h>
#include <hip/hip_bf16.h>
#include <math.h>

#define N_NODE   100000
#define N_EDGE   1000000
#define HID      64
#define ATTN     32
#define REL_VOCAB 2005
#define N_LAYER  2

typedef __attribute__((ext_vector_type(16))) _Float16 v16h;
typedef __attribute__((ext_vector_type(4)))  _Float16 v4h;
typedef __attribute__((ext_vector_type(8)))  float    v8f;

__device__ __forceinline__ v8f vzero8() {
  v8f z = {0.f,0.f,0.f,0.f,0.f,0.f,0.f,0.f};
  return z;
}

__device__ __forceinline__ v8f wmma_f16(v16h a, v16h b, v8f c) {
  // D = A(16x32 f16) * B(32x16 f16) + C(16x16 f32)
  return __builtin_amdgcn_wmma_f32_16x16x32_f16(
      /*neg_a=*/false, a, /*neg_b=*/false, b,
      /*c_mod=*/(short)0, c, /*reuse_a=*/false, /*reuse_b=*/false);
}

__device__ __forceinline__ float sigmoidf_(float x) {
  return 1.0f / (1.0f + __expf(-x));
}

// A-fragment (16x32 f16) from a row-major f32 matrix [rows][64]:
// lane holds row m=lane&15; VGPR v<4 -> K=2v, v>=4 -> K=8+2v; hi half +8.
__device__ __forceinline__ v16h load_afrag_g(const float* __restrict__ src,
                                             int row, int c, int hi) {
  v16h a;
#pragma unroll
  for (int v = 0; v < 8; ++v) {
    int k = c * 32 + (v < 4 ? 2 * v : 8 + 2 * v) + (hi ? 8 : 0);
    float2 f = *(const float2*)(src + (size_t)row * 64 + k);
    a[2 * v]     = (_Float16)f.x;
    a[2 * v + 1] = (_Float16)f.y;
  }
  return a;
}

// B-fragment (32x16 f16) for W^T from f16 LDS tile laid out [n][64]:
// lane holds column n; VGPR v<4 -> K=2v, v>=4 -> K=8+2v; hi half +8.
__device__ __forceinline__ v16h load_bfrag_lds(const _Float16 (*sw)[64],
                                               int n, int c, int hi) {
  v16h b;
#pragma unroll
  for (int v = 0; v < 8; ++v) {
    int k = c * 32 + (v < 4 ? 2 * v : 8 + 2 * v) + (hi ? 8 : 0);
    b[2 * v]     = sw[n][k];
    b[2 * v + 1] = sw[n][k + 1];
  }
  return b;
}

// ---------------------------------------------------------------------------
// Edge kernel: per block = 128 edges (8 waves x 16-edge M-tiles).
// a = relu([hs|hr] @ [Ws|Wr]^T) via WMMA; alpha = sigmoid(a @ wa + b);
// atomicAdd alpha*(hs+hr) into agg[obj].  All tables are L2-resident.
// ---------------------------------------------------------------------------
__global__ __launch_bounds__(256) void MASGNN_edge_kernel(
    const int*   __restrict__ edges,
    const float* __restrict__ rela,     // [REL_VOCAB][64] (layer slice)
    const float* __restrict__ Ws,       // [32][64]
    const float* __restrict__ Wr,       // [32][64]
    const float* __restrict__ wa,       // [32]
    const float* __restrict__ wb,       // [1]
    const float* __restrict__ hidden,   // [N_NODE][64]
    float*       __restrict__ agg)      // [N_NODE][64]
{
  __shared__ _Float16 s_cat[8][16][128];  // [hs|hr] f16 per wave tile
  __shared__ float    s_sum[8][16][64];   // hs+hr f32
  __shared__ _Float16 s_wT[32][128];      // combined [Ws|Wr], [n][k]
  __shared__ float    s_alpha[8][16];
  __shared__ int      s_sub[8][16], s_rel[8][16], s_obj[8][16];
  __shared__ float    s_wa[32];

  const int tid = threadIdx.x;
  const long long ebase = (long long)blockIdx.x * 128;

  // edge metadata (clamped for tail)
  for (int idx = tid; idx < 128; idx += 256) {
    long long ge = ebase + idx;
    int sub = 0, rel = 0, obj = 0;
    if (ge < N_EDGE) {
      sub = edges[ge * 6 + 4];
      rel = edges[ge * 6 + 2];
      obj = edges[ge * 6 + 5];
    }
    s_sub[idx >> 4][idx & 15] = sub;
    s_rel[idx >> 4][idx & 15] = rel;
    s_obj[idx >> 4][idx & 15] = obj;
  }
  // stage combined weight [Ws|Wr] as f16 [n][k] (float4 reads, packed stores)
  for (int idx = tid; idx < 32 * 32; idx += 256) {   // 32 n x 32 float4
    int n = idx >> 5, q = idx & 31;
    int k = q * 4;
    float4 f = (k < 64) ? *((const float4*)(Ws + n * 64) + q)
                        : *((const float4*)(Wr + n * 64) + (q - 16));
    v4h h4;
    h4[0] = (_Float16)f.x; h4[1] = (_Float16)f.y;
    h4[2] = (_Float16)f.z; h4[3] = (_Float16)f.w;
    *(v4h*)&s_wT[n][k] = h4;
  }
  if (tid < 32) s_wa[tid] = wa[tid];
  __syncthreads();

  // gather hs (hidden[sub]) and hr (rela[rel]) -> LDS, float4-wide
  for (int idx = tid; idx < 128 * 16; idx += 256) {
    int e = idx >> 4, q = idx & 15, w = e >> 4, m = e & 15;
    int h = q * 4;
    float4 vs = *((const float4*)(hidden + (size_t)s_sub[w][m] * 64) + q);
    float4 vr = *((const float4*)(rela   + (size_t)s_rel[w][m] * 64) + q);
    v4h hs4, hr4;
    hs4[0] = (_Float16)vs.x; hs4[1] = (_Float16)vs.y;
    hs4[2] = (_Float16)vs.z; hs4[3] = (_Float16)vs.w;
    hr4[0] = (_Float16)vr.x; hr4[1] = (_Float16)vr.y;
    hr4[2] = (_Float16)vr.z; hr4[3] = (_Float16)vr.w;
    *(v4h*)&s_cat[w][m][h]      = hs4;
    *(v4h*)&s_cat[w][m][64 + h] = hr4;
    float4 sm;
    sm.x = vs.x + vr.x; sm.y = vs.y + vr.y;
    sm.z = vs.z + vr.z; sm.w = vs.w + vr.w;
    *(float4*)&s_sum[w][m][h] = sm;
  }
  __syncthreads();

  const int w    = tid >> 5;
  const int lane = tid & 31;
  const int m    = lane & 15;
  const int hi   = (lane >= 16) ? 1 : 0;

  v8f acc0 = vzero8(), acc1 = vzero8();
#pragma unroll
  for (int c = 0; c < 4; ++c) {       // K=128 in 4 chunks of 32
    const int kb = c * 32;
    v16h a, b0, b1;
#pragma unroll
    for (int v = 0; v < 8; ++v) {
      int ka = kb + (v < 4 ? 2 * v : 8 + 2 * v) + (hi ? 8 : 0);
      a[2 * v]     = s_cat[w][m][ka];
      a[2 * v + 1] = s_cat[w][m][ka + 1];
      int kq = kb + (v < 4 ? 2 * v : 8 + 2 * v) + (hi ? 8 : 0);
      b0[2 * v]     = s_wT[m][kq];
      b0[2 * v + 1] = s_wT[m][kq + 1];
      b1[2 * v]     = s_wT[16 + m][kq];
      b1[2 * v + 1] = s_wT[16 + m][kq + 1];
    }
    acc0 = wmma_f16(a, b0, acc0);     // attn cols 0..15
    acc1 = wmma_f16(a, b1, acc1);     // attn cols 16..31
  }

  // relu + per-row dot with w_alpha (row-wise reduce across 16 lanes)
  float wa0 = s_wa[m], wa1 = s_wa[16 + m];
  float p[8];
#pragma unroll
  for (int r = 0; r < 8; ++r) {
    float x0 = fmaxf(acc0[r], 0.0f);
    float x1 = fmaxf(acc1[r], 0.0f);
    p[r] = x0 * wa0 + x1 * wa1;
  }
#pragma unroll
  for (int r = 0; r < 8; ++r) {
#pragma unroll
    for (int msk = 1; msk < 16; msk <<= 1)
      p[r] += __shfl_xor(p[r], msk, 16);
  }
  float bias = wb[0];
  if (m == 0) {
#pragma unroll
    for (int r = 0; r < 8; ++r)
      s_alpha[w][r + hi * 8] = sigmoidf_(p[r] + bias);
  }
  __syncthreads();

  // scatter: agg[obj] += alpha * (hs + hr)   (L2-resident float atomics)
  for (int idx = tid; idx < 128 * 64; idx += 256) {
    int e = idx >> 6, h = idx & 63, ww = e >> 4, mm = e & 15;
    long long ge = ebase + e;
    if (ge < N_EDGE) {
      float msg = s_alpha[ww][mm] * s_sum[ww][mm][h];
      atomicAdd(&agg[(size_t)s_obj[ww][mm] * 64 + h], msg);
    }
  }
}

// ---------------------------------------------------------------------------
// Node kernel: per block = 128 nodes (8 waves x 16-node tiles).
// x = relu(agg @ Wh^T); h = GRU(x, h0); hidden <- h (hidden doubles as h0).
// Weights (Wh|Wih|Whh) staged once per block in LDS as f16 [n][k].
// ---------------------------------------------------------------------------
#define WROW_WH  0
#define WROW_WIH 64
#define WROW_WHH 256
#define WROWS    448   // 64 + 192 + 192

__global__ __launch_bounds__(256) void MASGNN_node_kernel(
    const float* __restrict__ agg,
    const float* __restrict__ Wh,    // [64][64]
    const float* __restrict__ Wih,   // [192][64]
    const float* __restrict__ Whh,   // [192][64]
    const float* __restrict__ bih,   // [192]
    const float* __restrict__ bhh,   // [192]
    float*       __restrict__ hidden)
{
  __shared__ _Float16 s_w[WROWS][64];   // 56 KB: all weights, f16 [n][k]
  __shared__ _Float16 s_x[8][16][64];   // GRU input x per wave tile

  const int tid  = threadIdx.x;
  const int w    = tid >> 5;
  const int lane = tid & 31;
  const int m    = lane & 15;
  const int hi   = (lane >= 16) ? 1 : 0;
  const int nb   = blockIdx.x * 128 + w * 16;

  // stage weights: 448 rows x 16 float4 = 7168 float4s
  for (int idx = tid; idx < WROWS * 16; idx += 256) {
    int n = idx >> 4, q = idx & 15;
    const float* src;
    if (n < WROW_WIH)       src = Wh  + (size_t)n * 64;
    else if (n < WROW_WHH)  src = Wih + (size_t)(n - WROW_WIH) * 64;
    else                    src = Whh + (size_t)(n - WROW_WHH) * 64;
    float4 f = *((const float4*)src + q);
    v4h h4;
    h4[0] = (_Float16)f.x; h4[1] = (_Float16)f.y;
    h4[2] = (_Float16)f.z; h4[3] = (_Float16)f.w;
    *(v4h*)&s_w[n][q * 4] = h4;
  }

  int rowA = nb + m;
  if (rowA >= N_NODE) rowA = N_NODE - 1;

  // A-fragments from agg (rows are contiguous nodes -> straight global loads)
  v16h aA0 = load_afrag_g(agg, rowA, 0, hi);
  v16h aA1 = load_afrag_g(agg, rowA, 1, hi);
  __syncthreads();

  // --- Phase 1: x = relu(agg @ Wh^T) ---
#pragma unroll
  for (int t = 0; t < 4; ++t) {
    v8f acc = vzero8();
    acc = wmma_f16(aA0, load_bfrag_lds(s_w, WROW_WH + t * 16 + m, 0, hi), acc);
    acc = wmma_f16(aA1, load_bfrag_lds(s_w, WROW_WH + t * 16 + m, 1, hi), acc);
#pragma unroll
    for (int r = 0; r < 8; ++r)
      s_x[w][r + hi * 8][t * 16 + m] = (_Float16)fmaxf(acc[r], 0.0f);
  }
  __syncthreads();

  // --- Phase 2: GRU ---
  v16h xA0, xA1;
#pragma unroll
  for (int v = 0; v < 8; ++v) {
    int k0 = (v < 4 ? 2 * v : 8 + 2 * v) + (hi ? 8 : 0);
    xA0[2 * v]     = s_x[w][m][k0];
    xA0[2 * v + 1] = s_x[w][m][k0 + 1];
    xA1[2 * v]     = s_x[w][m][32 + k0];
    xA1[2 * v + 1] = s_x[w][m][32 + k0 + 1];
  }
  v16h hA0 = load_afrag_g(hidden, rowA, 0, hi);
  v16h hA1 = load_afrag_g(hidden, rowA, 1, hi);

#pragma unroll
  for (int t = 0; t < 4; ++t) {
    const int col = t * 16 + m;
    v8f gir = vzero8(), giz = vzero8(), gin = vzero8();
    v8f ghr = vzero8(), ghz = vzero8(), ghn = vzero8();

    gir = wmma_f16(xA0, load_bfrag_lds(s_w, WROW_WIH + col,       0, hi), gir);
    gir = wmma_f16(xA1, load_bfrag_lds(s_w, WROW_WIH + col,       1, hi), gir);
    giz = wmma_f16(xA0, load_bfrag_lds(s_w, WROW_WIH + 64 + col,  0, hi), giz);
    giz = wmma_f16(xA1, load_bfrag_lds(s_w, WROW_WIH + 64 + col,  1, hi), giz);
    gin = wmma_f16(xA0, load_bfrag_lds(s_w, WROW_WIH + 128 + col, 0, hi), gin);
    gin = wmma_f16(xA1, load_bfrag_lds(s_w, WROW_WIH + 128 + col, 1, hi), gin);
    ghr = wmma_f16(hA0, load_bfrag_lds(s_w, WROW_WHH + col,       0, hi), ghr);
    ghr = wmma_f16(hA1, load_bfrag_lds(s_w, WROW_WHH + col,       1, hi), ghr);
    ghz = wmma_f16(hA0, load_bfrag_lds(s_w, WROW_WHH + 64 + col,  0, hi), ghz);
    ghz = wmma_f16(hA1, load_bfrag_lds(s_w, WROW_WHH + 64 + col,  1, hi), ghz);
    ghn = wmma_f16(hA0, load_bfrag_lds(s_w, WROW_WHH + 128 + col, 0, hi), ghn);
    ghn = wmma_f16(hA1, load_bfrag_lds(s_w, WROW_WHH + 128 + col, 1, hi), ghn);

    float b_ir = bih[col], b_iz = bih[64 + col], b_in = bih[128 + col];
    float b_hr = bhh[col], b_hz = bhh[64 + col], b_hn = bhh[128 + col];
#pragma unroll
    for (int r = 0; r < 8; ++r) {
      int row  = nb + r + hi * 8;
      int rowc = (row < N_NODE) ? row : (N_NODE - 1);
      float hold = hidden[(size_t)rowc * 64 + col];
      float rg = sigmoidf_((gir[r] + b_ir) + (ghr[r] + b_hr));
      float zg = sigmoidf_((giz[r] + b_iz) + (ghz[r] + b_hz));
      float ng = tanhf((gin[r] + b_in) + rg * (ghn[r] + b_hn));
      float hnew = (1.0f - zg) * ng + zg * hold;
      if (row < N_NODE) hidden[(size_t)row * 64 + col] = hnew;
    }
  }
}

__global__ void MASGNN_zero_kernel(float* __restrict__ p, long long n) {
  long long i = (long long)blockIdx.x * blockDim.x + threadIdx.x;
  if (i < n) p[i] = 0.0f;
}

__global__ void MASGNN_final_kernel(const float* __restrict__ hidden,
                                    const float* __restrict__ Wf,
                                    float* __restrict__ out) {
  int i = blockIdx.x * blockDim.x + threadIdx.x;
  if (i < N_NODE) {
    const float4* h4 = (const float4*)(hidden + (size_t)i * 64);
    const float4* w4 = (const float4*)Wf;
    float s = 0.0f;
#pragma unroll
    for (int q = 0; q < 16; ++q) {
      float4 h = h4[q], w = w4[q];
      s += h.x * w.x + h.y * w.y + h.z * w.z + h.w * w.w;
    }
    out[i] = s;
  }
}

extern "C" void kernel_launch(void* const* d_in, const int* in_sizes, int n_in,
                              void* d_out, int out_size, void* d_ws, size_t ws_size,
                              hipStream_t stream) {
  (void)in_sizes; (void)n_in; (void)out_size; (void)ws_size;

  const int*   edges = (const int*)d_in[0];
  const float* rela  = (const float*)d_in[1];
  const float* Ws    = (const float*)d_in[2];
  const float* Wr    = (const float*)d_in[3];
  const float* waw   = (const float*)d_in[4];
  const float* wab   = (const float*)d_in[5];
  const float* Wh    = (const float*)d_in[6];
  const float* Wih   = (const float*)d_in[7];
  const float* Whh   = (const float*)d_in[8];
  const float* bih   = (const float*)d_in[9];
  const float* bhh   = (const float*)d_in[10];
  const float* Wf    = (const float*)d_in[11];
  float* out = (float*)d_out;

  // workspace: hidden (doubles as h0, since hidden==h0 after every layer) + agg
  float* hidden = (float*)d_ws;
  float* agg    = hidden + (size_t)N_NODE * HID;

  const long long nstate = (long long)N_NODE * HID;
  const int zgrid = (int)((nstate + 255) / 256);
  MASGNN_zero_kernel<<<zgrid, 256, 0, stream>>>(hidden, nstate);

  const int egrid = (N_EDGE + 127) / 128;
  const int ngrid = (N_NODE + 127) / 128;

  for (int l = 0; l < N_LAYER; ++l) {
    MASGNN_zero_kernel<<<zgrid, 256, 0, stream>>>(agg, nstate);
    MASGNN_edge_kernel<<<egrid, 256, 0, stream>>>(
        edges,
        rela + (size_t)l * REL_VOCAB * HID,
        Ws + (size_t)l * ATTN * HID,
        Wr + (size_t)l * ATTN * HID,
        waw + (size_t)l * ATTN,
        wab + l,
        hidden, agg);
    MASGNN_node_kernel<<<ngrid, 256, 0, stream>>>(
        agg,
        Wh  + (size_t)l * HID * HID,
        Wih + (size_t)l * 3 * HID * HID,
        Whh + (size_t)l * 3 * HID * HID,
        bih + (size_t)l * 3 * HID,
        bhh + (size_t)l * 3 * HID,
        hidden);
  }

  MASGNN_final_kernel<<<(N_NODE + 255) / 256, 256, 0, stream>>>(hidden, Wf, out);
}
